// GATSeqClassifier_11364483465674
// MI455X (gfx1250) — matrix operand order repaired and checked by reading, not compiled, not run on hardware
//
#include <hip/hip_runtime.h>
#include <hip/hip_bf16.h>
#include <stdint.h>

// ---------------------------------------------------------------------------
// GATv2Conv for MI455X (gfx1250, wave32).
//   N=50000 nodes, E=800000 edges, F=128 in, H=4 heads, C=32 -> H*C=128.
// Pipeline:
//   k0: init out=bias, m_enc=0, den=0
//   k1: x_l = x@W_l^T, x_r = x@W_r^T via v_wmma_f32_16x16x32_bf16
//       (x tile + both weight matrices staged in LDS as bf16; 102KB LDS —
//        uses CDNA5's 320KB/WGP capacity)
//   k2: per-edge logits (LeakyReLU + att dot) + atomic segment max
//   k3: num = exp(logit - m), atomic segment sum den
//   k4: out[dst] += alpha * x_l[src]  (f32 global atomics)
// ---------------------------------------------------------------------------

typedef __attribute__((ext_vector_type(16))) __bf16 v16bf;
typedef __attribute__((ext_vector_type(8)))  float  v8f;

#define FDIM 128
#define HEADS 4
#define CPH 32
#define NEG_SLOPE 0.2f

union Frag16 {
    v16bf v;
    unsigned short s[16];
    uint4 q[2];
};

__device__ __forceinline__ unsigned short f2bf(float f) {
    unsigned u = __float_as_uint(f);
    unsigned r = u + 0x7FFFu + ((u >> 16) & 1u);  // round-to-nearest-even
    return (unsigned short)(r >> 16);
}

__device__ __forceinline__ ushort4 f2bf4(float4 v) {
    ushort4 b;
    b.x = f2bf(v.x); b.y = f2bf(v.y); b.z = f2bf(v.z); b.w = f2bf(v.w);
    return b;
}

// order-preserving float -> uint map for atomicMax-based segment max
__device__ __forceinline__ unsigned enc_f(float f) {
    unsigned u = __float_as_uint(f);
    return (u & 0x80000000u) ? ~u : (u | 0x80000000u);
}
__device__ __forceinline__ float dec_u(unsigned e) {
    unsigned u = (e & 0x80000000u) ? (e & 0x7FFFFFFFu) : ~e;
    return __uint_as_float(u);
}

// ---------------------------------------------------------------------------
// k0: init out with bias, clear m_enc / den
// ---------------------------------------------------------------------------
__global__ __launch_bounds__(256) void k_init(float* __restrict__ out,
                                              const float* __restrict__ bias,
                                              unsigned* __restrict__ m_enc,
                                              float* __restrict__ den,
                                              int n) {
    int idx = blockIdx.x * 256 + threadIdx.x;
    int total = n * FDIM;
    if (idx < total) out[idx] = bias[idx & (FDIM - 1)];
    if (idx < n * HEADS) { m_enc[idx] = 0u; den[idx] = 0.0f; }
}

// ---------------------------------------------------------------------------
// k1: dual GEMM  x_l = x@W_l^T, x_r = x@W_r^T   (bf16 WMMA, f32 accumulate)
// block = 256 threads = 8 waves; x tile + W_l + W_r all bf16 in LDS.
// waves 0-3 -> W_l/x_l, waves 4-7 -> W_r/x_r; each wave owns 32 rows.
// ---------------------------------------------------------------------------
#define XSTR 136  // 128 + 8 bf16 pad: 272B row stride -> 4-bank rotation/row

__global__ __launch_bounds__(256) void k_gemm(const float* __restrict__ x,
                                              const float* __restrict__ Wl,
                                              const float* __restrict__ Wr,
                                              float* __restrict__ xl,
                                              float* __restrict__ xr,
                                              int nrows) {
    __shared__ unsigned short sx [128 * XSTR];
    __shared__ unsigned short swl[128 * XSTR];
    __shared__ unsigned short swr[128 * XSTR];

    const int t = threadIdx.x;
    const int row0 = blockIdx.x * 128;
    const bool full = (row0 + 128 <= nrows);   // block-uniform

    // ---- cooperative fill: x tile, W_l, W_r  (f32 -> bf16) ----
    {
        const int c4 = (t & 31) * 4;   // column group (float4)
        const int r0 = t >> 5;         // 0..7
#pragma unroll
        for (int i = 0; i < 16; ++i) {
            const int lr = r0 + i * 8;
            const int gr = row0 + lr;
            float4 v = make_float4(0.f, 0.f, 0.f, 0.f);
            if (full || gr < nrows)
                v = *(const float4*)(x + (size_t)gr * FDIM + c4);
            *(ushort4*)(&sx[lr * XSTR + c4]) = f2bf4(v);

            float4 wv = *(const float4*)(Wl + (size_t)lr * FDIM + c4);
            *(ushort4*)(&swl[lr * XSTR + c4]) = f2bf4(wv);
            float4 uv = *(const float4*)(Wr + (size_t)lr * FDIM + c4);
            *(ushort4*)(&swr[lr * XSTR + c4]) = f2bf4(uv);
        }
    }
    __syncthreads();

    const int wave = t >> 5;
    const int lane = t & 31;
    const int lr15 = lane & 15;
    const int lhi  = lane >> 4;       // 0 or 1

    const unsigned short* sW  = (wave < 4) ? swl : swr;
    float*                dst = (wave < 4) ? xl : xr;
    const int wm = (wave & 3) * 32;   // this wave's 32-row slice of the tile

    // prefetch next block's x tile rows while we compute (global_prefetch_b8)
    if (row0 + 128 + (t >> 1) < nrows)
        __builtin_prefetch(x + (size_t)(row0 + 128 + (t >> 1)) * FDIM, 0, 1);

    for (int mt = 0; mt < 2; ++mt) {
        const int m0 = wm + mt * 16;
        const int arow = m0 + lr15;

        // A fragments, all 4 K-chunks (ISA 16-bit A 16x32 layout):
        // lane<16: K = kc+0..7 and kc+16..23 ; lane>=16: +8
        Frag16 A[4];
#pragma unroll
        for (int kc = 0; kc < 4; ++kc) {
            const int klo = kc * 32 + lhi * 8;
            A[kc].q[0] = *(const uint4*)(&sx[arow * XSTR + klo]);
            A[kc].q[1] = *(const uint4*)(&sx[arow * XSTR + klo + 16]);
        }

#pragma unroll
        for (int nt = 0; nt < 8; ++nt) {
            const int ocol = nt * 16 + lr15;          // output channel = B col
            const unsigned short* wrow = sW + (size_t)ocol * XSTR;

            v8f acc = {};
#pragma unroll
            for (int kc = 0; kc < 4; ++kc) {
                // B 32x16: lane<16 holds K=kc+0..7 and kc+16..23, lane>=16: +8;
                // two 16B chunks in the W row -> two ds_load_b128
                const int kb = kc * 32 + lhi * 8;
                Frag16 B;
                B.q[0] = *(const uint4*)(wrow + kb);
                B.q[1] = *(const uint4*)(wrow + kb + 16);
                acc = __builtin_amdgcn_wmma_f32_16x16x32_bf16(
                    false, A[kc].v, false, B.v, (short)0, acc, false, false);
            }

            // D layout: VGPR r -> M = m0 + r + 8*lhi, N = lr15 (+nt*16)
            float* dptr = dst + (size_t)(row0 + m0 + lhi * 8) * FDIM + ocol;
            if (full) {
#pragma unroll
                for (int r = 0; r < 8; ++r)
                    dptr[(size_t)r * FDIM] = acc[r];
            } else {
#pragma unroll
                for (int r = 0; r < 8; ++r) {
                    if (row0 + m0 + lhi * 8 + r < nrows)
                        dptr[(size_t)r * FDIM] = acc[r];
                }
            }
        }
    }
}

// ---------------------------------------------------------------------------
// k2: per-edge attention logit + atomic segment max.
// one wave per edge; lane c owns channels 4c..4c+3 (head = lane>>3).
// ---------------------------------------------------------------------------
__global__ __launch_bounds__(256) void k_logits(const long long* __restrict__ ei,
                                                const float* __restrict__ xl,
                                                const float* __restrict__ xr,
                                                const float* __restrict__ att,
                                                float* __restrict__ logits,
                                                unsigned* __restrict__ m_enc,
                                                int E, int Et) {
    const int e = blockIdx.x * 8 + (threadIdx.x >> 5);
    if (e >= Et) return;
    const int lane = threadIdx.x & 31;

    int src, dst;
    if (e < E) { src = (int)ei[e]; dst = (int)ei[E + e]; }
    else       { src = e - E; dst = e - E; }  // self loop

    const int c4 = lane * 4;
    const int h  = lane >> 3;
    float4 a = *(const float4*)(xl + (size_t)src * FDIM + c4);
    float4 b = *(const float4*)(xr + (size_t)dst * FDIM + c4);
    float4 w = *(const float4*)(att + h * CPH + (lane & 7) * 4);

    float sx = a.x + b.x, sy = a.y + b.y, sz = a.z + b.z, sw = a.w + b.w;
    sx = (sx > 0.f) ? sx : NEG_SLOPE * sx;
    sy = (sy > 0.f) ? sy : NEG_SLOPE * sy;
    sz = (sz > 0.f) ? sz : NEG_SLOPE * sz;
    sw = (sw > 0.f) ? sw : NEG_SLOPE * sw;
    float s = sx * w.x + sy * w.y + sz * w.z + sw * w.w;

    // reduce over the 8 lanes of this head
    s += __shfl_xor(s, 1, 32);
    s += __shfl_xor(s, 2, 32);
    s += __shfl_xor(s, 4, 32);

    if ((lane & 7) == 0) {
        logits[(size_t)e * HEADS + h] = s;
        atomicMax(&m_enc[dst * HEADS + h], enc_f(s));
    }
}

// ---------------------------------------------------------------------------
// k3: num = exp(logit - m[dst]); den[dst] += num  (in-place over logits)
// ---------------------------------------------------------------------------
__global__ __launch_bounds__(256) void k_expsum(const long long* __restrict__ ei,
                                                float* __restrict__ logits,
                                                const unsigned* __restrict__ m_enc,
                                                float* __restrict__ den,
                                                int E, int Et) {
    const int idx = blockIdx.x * 256 + threadIdx.x;
    if (idx >= Et * HEADS) return;
    const int e = idx >> 2;
    const int h = idx & 3;
    const int dst = (e < E) ? (int)ei[E + e] : (e - E);
    const float m = dec_u(m_enc[dst * HEADS + h]);
    const float num = __expf(logits[idx] - m);
    logits[idx] = num;
    atomicAdd(&den[dst * HEADS + h], num);
}

// ---------------------------------------------------------------------------
// k4: out[dst] += alpha * x_l[src]; one wave per edge, float4 per lane.
// ---------------------------------------------------------------------------
__global__ __launch_bounds__(256) void k_aggregate(const long long* __restrict__ ei,
                                                   const float* __restrict__ xl,
                                                   const float* __restrict__ num,
                                                   const float* __restrict__ den,
                                                   float* __restrict__ out,
                                                   int E, int Et) {
    const int e = blockIdx.x * 8 + (threadIdx.x >> 5);
    if (e >= Et) return;
    const int lane = threadIdx.x & 31;

    int src, dst;
    if (e < E) { src = (int)ei[e]; dst = (int)ei[E + e]; }
    else       { src = e - E; dst = e - E; }

    const int h = lane >> 3;
    const float d = den[dst * HEADS + h];
    const float alpha = num[(size_t)e * HEADS + h] / fmaxf(d, 1e-16f);

    const int c4 = lane * 4;
    float4 v = *(const float4*)(xl + (size_t)src * FDIM + c4);
    float* o = out + (size_t)dst * FDIM + c4;
    atomicAdd(o + 0, alpha * v.x);
    atomicAdd(o + 1, alpha * v.y);
    atomicAdd(o + 2, alpha * v.z);
    atomicAdd(o + 3, alpha * v.w);
}

// ---------------------------------------------------------------------------
extern "C" void kernel_launch(void* const* d_in, const int* in_sizes, int n_in,
                              void* d_out, int out_size, void* d_ws, size_t ws_size,
                              hipStream_t stream) {
    const float*     x    = (const float*)d_in[0];
    const long long* ei   = (const long long*)d_in[1];  // int64 edge_index [2,E]
    const float*     Wl   = (const float*)d_in[2];
    const float*     Wr   = (const float*)d_in[3];
    const float*     att  = (const float*)d_in[4];
    const float*     bias = (const float*)d_in[5];
    float*           out  = (float*)d_out;

    const int n  = in_sizes[0] / FDIM;      // 50000
    const int E  = in_sizes[1] / 2;         // 800000
    const int Et = E + n;                   // + self loops

    // workspace carve-up (floats)
    float*    xl     = (float*)d_ws;
    float*    xr     = xl + (size_t)n * FDIM;
    float*    logits = xr + (size_t)n * FDIM;          // reused as `num`
    unsigned* m_enc  = (unsigned*)(logits + (size_t)Et * HEADS);
    float*    den    = (float*)(m_enc + (size_t)n * HEADS);

    // k0: init
    {
        int total = n * FDIM;
        k_init<<<(total + 255) / 256, 256, 0, stream>>>(out, bias, m_enc, den, n);
    }
    // k1: projections via WMMA
    k_gemm<<<(n + 127) / 128, 256, 0, stream>>>(x, Wl, Wr, xl, xr, n);
    // k2: logits + segment max
    k_logits<<<(Et + 7) / 8, 256, 0, stream>>>(ei, xl, xr, att, logits, m_enc, E, Et);
    // k3: exp + segment sum
    k_expsum<<<(Et * HEADS + 255) / 256, 256, 0, stream>>>(ei, logits, m_enc, den, E, Et);
    // k4: weighted scatter aggregation
    k_aggregate<<<(Et + 7) / 8, 256, 0, stream>>>(ei, xl, logits, den, out, E, Et);
}
